// Attention4D_65180423685452
// MI455X (gfx1250) — matrix operand order repaired and checked by reading, not compiled, not run on hardware
//
#include <hip/hip_runtime.h>
#include <hip/hip_bf16.h>

// ---------- types ----------
typedef __attribute__((ext_vector_type(16))) _Float16 v16h;
typedef __attribute__((ext_vector_type(8)))  _Float16 v8h;
typedef __attribute__((ext_vector_type(8)))  float    v8f;

#define HEADS 8
#define KD    32
#define RES   28
#define NTOK  784          // RES*RES
#define DIM   384
#define DHEAD 128          // AR*KD
#define DH    1024         // HEADS*DHEAD
#define NH_KD 256          // HEADS*KD
#define BATCH 8

// ---------- WMMA helpers ----------
__device__ __forceinline__ v8f wmma_f16(v16h a, v16h b, v8f c) {
  return __builtin_amdgcn_wmma_f32_16x16x32_f16(
      /*neg_a=*/false, a, /*neg_b=*/false, b,
      /*c_mod=*/(short)0, c, /*reuse_a=*/false, /*reuse_b=*/false);
}

__device__ __forceinline__ v16h combine(v8h lo, v8h hi) {
  return __builtin_shufflevector(lo, hi, 0,1,2,3,4,5,6,7,8,9,10,11,12,13,14,15);
}

// A-fragment (16x32 f16) from an f16 row (row-major along K, this lane's row).
// halves j: K = k0 + j + (lane>=16 ? 16 : 0)
__device__ __forceinline__ v16h load_afrag_f16(const _Float16* __restrict__ row,
                                               int k0, int lane) {
  int koff = k0 + ((lane & 16) ? 16 : 0);
  v8h lo = *(const v8h*)(row + koff);
  v8h hi = *(const v8h*)(row + koff + 8);
  return combine(lo, hi);
}

// B-fragment (32x16 f16): per-lane 16 contiguous halves; caller passes the
// lane's column base pointer already including the (lane>=16 ? +16 : 0) K skew.
__device__ __forceinline__ v16h load_bfrag_f16(const _Float16* __restrict__ colrow,
                                               int k0) {
  v8h lo = *(const v8h*)(colrow + k0);
  v8h hi = *(const v8h*)(colrow + k0 + 8);
  return combine(lo, hi);
}

// ---------- kernel W: one-shot weight convert to f16, BN scale folded ----------
// wh  : [1536][384] f16  (rows 0..255 = Wq*qs, 256..511 = Wk*ks, 512..1535 = Wv*vs)
// wph : [384][1024] f16  (Wp * ps)
__global__ __launch_bounds__(256) void k_wcvt(
    const float* __restrict__ Wq, const float* __restrict__ qs,
    const float* __restrict__ Wk, const float* __restrict__ ks,
    const float* __restrict__ Wv, const float* __restrict__ vs,
    const float* __restrict__ Wp, const float* __restrict__ ps,
    _Float16* __restrict__ wh, _Float16* __restrict__ wph) {
  int i = blockIdx.x * 256 + threadIdx.x;
  if (i < 1536 * DIM) {
    int o = i / DIM;
    float s, w;
    if (o < NH_KD)          { s = qs[o];            w = Wq[i]; }
    else if (o < 2 * NH_KD) { s = ks[o - NH_KD];    w = Wk[i - NH_KD * DIM]; }
    else                    { s = vs[o - 2 * NH_KD]; w = Wv[i - 2 * NH_KD * DIM]; }
    wh[i] = (_Float16)(w * s);
  } else {
    int j = i - 1536 * DIM;
    if (j < DIM * DH) {
      int o = j / DH;
      wph[j] = (_Float16)(Wp[j] * ps[o]);
    }
  }
}

// ---------- kernel B: one-shot folded biases (b*s + shift) ----------
__global__ __launch_bounds__(256) void k_bcvt(
    const float* __restrict__ bq, const float* __restrict__ qs, const float* __restrict__ qb,
    const float* __restrict__ bk, const float* __restrict__ ks, const float* __restrict__ kb,
    const float* __restrict__ bv, const float* __restrict__ vs, const float* __restrict__ vb,
    const float* __restrict__ bp, const float* __restrict__ ps, const float* __restrict__ pb,
    float* __restrict__ bias, float* __restrict__ biasp) {
  int i = blockIdx.x * 256 + threadIdx.x;
  if (i < 1536) {
    float s, bb, sh;
    if (i < NH_KD)          { s = qs[i];  bb = bq[i];  sh = qb[i]; }
    else if (i < 2 * NH_KD) { int l = i - NH_KD;     s = ks[l]; bb = bk[l]; sh = kb[l]; }
    else                    { int l = i - 2 * NH_KD; s = vs[l]; bb = bv[l]; sh = vb[l]; }
    bias[i] = bb * s + sh;
  } else if (i < 1536 + DIM) {
    int l = i - 1536;
    biasp[l] = bp[l] * ps[l] + pb[l];
  }
}

// ---------- kernel 0: x (B,384,784) f32 -> x_t (B,784,384) f16 ----------
__global__ __launch_bounds__(256) void k_xt(const float* __restrict__ x,
                                            _Float16* __restrict__ xt) {
  int i = blockIdx.x * 256 + threadIdx.x;
  if (i >= BATCH * DIM * NTOK) return;
  int n = i % NTOK;
  int c = (i / NTOK) % DIM;
  int b = i / (NTOK * DIM);
  xt[((size_t)b * NTOK + n) * DIM + c] = (_Float16)x[i];
}

// ---------- kernel 1: fused QKV projection GEMM (WMMA), BN folded ----------
// Y(o,n) = sum_c wh(o,c) * x(c,n) + bias(o)
// rows   0..255  -> q_t[b,h,n,kd]   (transposed f16)
// rows 256..511  -> k_t[b,h,m,kd]   (transposed f16)
// rows 512..1535 -> v_n[b,c,m]      (natural   f16)
__global__ __launch_bounds__(256) void k_qkv(
    const _Float16* __restrict__ xt, const _Float16* __restrict__ wh,
    const float* __restrict__ bias,
    _Float16* __restrict__ q_t, _Float16* __restrict__ k_t,
    _Float16* __restrict__ v_n) {
  int wave = (blockIdx.x * 256 + threadIdx.x) >> 5;
  int lane = threadIdx.x & 31;
  // tiles: b(8) x otile(96) x ntile(49)
  int nt = wave % 49;
  int ot = (wave / 49) % 96;
  int b  = wave / (49 * 96);
  int o0 = ot * 16, n0 = nt * 16;

  int col  = lane & 15;
  int roff = (lane & 16) ? 8 : 0;
  const _Float16* wrow = wh + (size_t)(o0 + col) * DIM;
  const _Float16* xrow = xt + ((size_t)b * NTOK + n0 + col) * DIM + ((lane & 16) ? 16 : 0);

  v8f acc = {};
#pragma unroll
  for (int k0 = 0; k0 < DIM; k0 += 32) {
    __builtin_prefetch(xrow + k0 + 32, 0, 1);
    v16h a  = load_afrag_f16(wrow, k0, lane);
    v16h bf = load_bfrag_f16(xrow, k0);
    acc = wmma_f16(a, bf, acc);
  }

  // epilogue: add folded bias, route to layout
  float y[8];
#pragma unroll
  for (int i = 0; i < 8; ++i) y[i] = acc[i] + bias[o0 + i + roff];

  if (o0 < 2 * NH_KD) {           // q or k: transposed (n, kd), 8 contiguous kd
    int oloc = (o0 < NH_KD) ? o0 : (o0 - NH_KD);
    int gr0  = oloc + roff;
    int hh   = gr0 >> 5;
    int kd0  = gr0 & 31;
    _Float16* dst = (o0 < NH_KD ? q_t : k_t) +
        (((size_t)b * HEADS + hh) * NTOK + n0 + col) * KD + kd0;
    v8h ov;
#pragma unroll
    for (int i = 0; i < 8; ++i) ov[i] = (_Float16)y[i];
    *(v8h*)dst = ov;
  } else {                        // v: natural (c, n)
    int oloc = o0 - 2 * NH_KD;
#pragma unroll
    for (int i = 0; i < 8; ++i) {
      int cloc = oloc + i + roff;
      v_n[((size_t)b * DH + cloc) * NTOK + n0 + col] = (_Float16)y[i];
    }
  }
}

// ---------- kernel 2: depthwise 3x3 conv branch -> vl (f32, natural) ----------
__global__ __launch_bounds__(256) void k_dw(const _Float16* __restrict__ v_n,
                                            const float* __restrict__ Wvl,
                                            const float* __restrict__ bvl,
                                            const float* __restrict__ vls,
                                            const float* __restrict__ vlb,
                                            float* __restrict__ vl) {
  int i = blockIdx.x * 256 + threadIdx.x;
  if (i >= BATCH * DH * NTOK) return;
  int xp = i % RES;
  int yp = (i / RES) % RES;
  int c  = (i / NTOK) % DH;
  int b  = i / (NTOK * DH);
  const _Float16* src = v_n + ((size_t)b * DH + c) * NTOK;
  const float*    w   = Wvl + (size_t)c * 9;
  float acc = bvl[c];                    // (conv + bvl) * s + shift
#pragma unroll
  for (int dy = -1; dy <= 1; ++dy) {
    int yy = yp + dy;
    if (yy < 0 || yy >= RES) continue;
#pragma unroll
    for (int dx = -1; dx <= 1; ++dx) {
      int xx = xp + dx;
      if (xx < 0 || xx >= RES) continue;
      acc += (float)src[yy * RES + xx] * w[(dy + 1) * 3 + (dx + 1)];
    }
  }
  vl[i] = acc * vls[c] + vlb[c];
}

// ---------- kernel 3: fused attention ----------
// One workgroup per (b, 16-query-row tile). 8 waves = 8 heads.
// Full logit strip for all heads lives in LDS: L[16][784][8] f16 (196 KB of the
// 320 KB WGP LDS). Phases: QK^T(WMMA)+bias -> th1 mix -> softmax -> th2 mix ->
// attn@V(WMMA) -> +vl, store transposed for the projection GEMM.
__global__ __launch_bounds__(256) void k_attn(
    const _Float16* __restrict__ q_t, const _Float16* __restrict__ k_t,
    const _Float16* __restrict__ v_n,
    const float* __restrict__ Wth1, const float* __restrict__ bth1,
    const float* __restrict__ Wth2, const float* __restrict__ bth2,
    const float* __restrict__ abias, const int* __restrict__ bidx,
    const float* __restrict__ vl, _Float16* __restrict__ osum_t) {
  extern __shared__ char smem[];
  _Float16* L    = (_Float16*)smem;                            // [16][784][8]
  float*    sinv = (float*)(smem + (size_t)16 * NTOK * 8 * 2); // [16][8]
  float*    w1   = sinv + 128;
  float*    w2   = w1 + 64;
  float*    c1   = w2 + 64;
  float*    c2   = c1 + 8;

  int t    = threadIdx.x;
  int lane = t & 31;
  int h    = t >> 5;
  int b    = blockIdx.x / 49;
  int nt   = blockIdx.x % 49;
  int n0   = nt * 16;
  int col  = lane & 15;
  int roff = (lane & 16) ? 8 : 0;

  if (t < 64) { w1[t] = Wth1[t]; w2[t] = Wth2[t]; }
  if (t < 8)  { c1[t] = bth1[t]; c2[t] = bth2[t]; }

  // ---- phase 1: logits = q @ k^T * scale + relpos bias (per head) ----
  const float scale = 0.17677669529663687f;  // 32^-0.5
  const _Float16* qrow = q_t + (((size_t)b * HEADS + h) * NTOK + n0 + col) * KD;
  v16h qa = load_afrag_f16(qrow, 0, lane);   // KD == 32: single K-step
  const float* abh = abias + (size_t)h * NTOK;
  for (int mt = 0; mt < 49; ++mt) {
    int m0 = mt * 16;
    const _Float16* krow = k_t + (((size_t)b * HEADS + h) * NTOK + m0 + col) * KD +
                           ((lane & 16) ? 16 : 0);
    v16h kb = load_bfrag_f16(krow, 0);
    v8f cacc = {};
    cacc = wmma_f16(qa, kb, cacc);
    int m = m0 + col;
#pragma unroll
    for (int i = 0; i < 8; ++i) {
      int n = n0 + i + roff;
      int idx = bidx[(size_t)n * NTOK + m];
      float v = cacc[i] * scale + abh[idx];
      L[((size_t)(i + roff) * NTOK + m) * 8 + h] = (_Float16)v;
    }
  }
  __syncthreads();

  // ---- phase 2: talking-heads 1 (8x8 mix across heads, in place) ----
  for (int p = t; p < 16 * NTOK; p += 256) {
    _Float16* pp = &L[(size_t)p * 8];
    v8h xv = *(v8h*)pp;
    float xf[8];
#pragma unroll
    for (int j = 0; j < 8; ++j) xf[j] = (float)xv[j];
    v8h yv;
#pragma unroll
    for (int g = 0; g < 8; ++g) {
      float y = c1[g];
#pragma unroll
      for (int j = 0; j < 8; ++j) y += w1[g * 8 + j] * xf[j];
      yv[g] = (_Float16)y;
    }
    *(v8h*)pp = yv;
  }
  __syncthreads();

  // ---- phase 3: softmax over m per (row, head); store unnormalized exp ----
  if (t < 128) {
    int r  = t >> 3;
    int hh = t & 7;
    _Float16* base = &L[(size_t)r * NTOK * 8 + hh];
    float mx = -3.0e30f;
    for (int m = 0; m < NTOK; ++m) {
      float v = (float)base[(size_t)m * 8];
      mx = fmaxf(mx, v);
    }
    float sum = 0.f;
    for (int m = 0; m < NTOK; ++m) {
      float e = __expf((float)base[(size_t)m * 8] - mx);
      base[(size_t)m * 8] = (_Float16)e;
      sum += e;
    }
    sinv[r * 8 + hh] = 1.0f / sum;
  }
  __syncthreads();

  // ---- phase 4: talking-heads 2 (normalize + 8x8 mix + bias, in place) ----
  for (int p = t; p < 16 * NTOK; p += 256) {
    int r = p / NTOK;
    _Float16* pp = &L[(size_t)p * 8];
    v8h xv = *(v8h*)pp;
    float xf[8];
#pragma unroll
    for (int j = 0; j < 8; ++j) xf[j] = (float)xv[j] * sinv[r * 8 + j];
    v8h yv;
#pragma unroll
    for (int g = 0; g < 8; ++g) {
      float y = c2[g];
#pragma unroll
      for (int j = 0; j < 8; ++j) y += w2[g * 8 + j] * xf[j];
      yv[g] = (_Float16)y;
    }
    *(v8h*)pp = yv;
  }
  __syncthreads();

  // ---- phase 5: o = attn @ v (16 x 128 per head), + vl, store transposed ----
  v8f acc[8] = {};
  int r = lane & 15;   // A-fragment row (both lane halves hold M = lane%16)
  for (int kt = 0; kt < 25; ++kt) {   // 784 = 24*32 + 16 (tail zero-padded)
    int m0 = kt * 32;
    v16h a;
    int mo = m0 + ((lane & 16) ? 16 : 0);
#pragma unroll
    for (int j = 0; j < 8; ++j) {
      int m = mo + j;
      a[j] = (m < NTOK) ? L[((size_t)r * NTOK + m) * 8 + h] : (_Float16)0.f;
    }
#pragma unroll
    for (int j = 0; j < 8; ++j) {
      int m = mo + 8 + j;
      a[j + 8] = (m < NTOK) ? L[((size_t)r * NTOK + m) * 8 + h] : (_Float16)0.f;
    }
#pragma unroll
    for (int dt = 0; dt < 8; ++dt) {
      int d = dt * 16 + col;
      const _Float16* vrow = v_n + ((size_t)b * DH + h * DHEAD + d) * NTOK;
      int mb = m0 + ((lane & 16) ? 16 : 0);
      v16h bf;
      if (mb + 15 < NTOK) {
        bf = load_bfrag_f16(vrow, mb);
      } else {
#pragma unroll
        for (int j = 0; j < 16; ++j) {
          int m = mb + j;
          bf[j] = (m < NTOK) ? vrow[m] : (_Float16)0.f;
        }
      }
      acc[dt] = wmma_f16(a, bf, acc[dt]);
    }
  }
  // epilogue: add dwconv branch, write osum_t[b, n, c] (f16)
#pragma unroll
  for (int dt = 0; dt < 8; ++dt) {
    int c = h * DHEAD + dt * 16 + col;
    const float* vlp = vl + ((size_t)b * DH + c) * NTOK + n0 + roff;
    _Float16* op = osum_t + ((size_t)b * NTOK + n0 + roff) * DH + c;
#pragma unroll
    for (int i = 0; i < 8; ++i) {
      float y = acc[dt][i] + vlp[i];
      op[(size_t)i * DH] = (_Float16)y;
    }
  }
}

// ---------- kernel 4: output projection GEMM (WMMA), BN folded ----------
__global__ __launch_bounds__(256) void k_proj(const _Float16* __restrict__ osum_t,
                                              const _Float16* __restrict__ wph,
                                              const float* __restrict__ biasp,
                                              float* __restrict__ out) {
  int wave = (blockIdx.x * 256 + threadIdx.x) >> 5;
  int lane = threadIdx.x & 31;
  // tiles: b(8) x otile(24) x ntile(49)
  int nt = wave % 49;
  int ot = (wave / 49) % 24;
  int b  = wave / (49 * 24);
  int o0 = ot * 16, n0 = nt * 16;
  int col  = lane & 15;
  int roff = (lane & 16) ? 8 : 0;

  const _Float16* wrow = wph + (size_t)(o0 + col) * DH;
  const _Float16* xrow = osum_t + ((size_t)b * NTOK + n0 + col) * DH +
                         ((lane & 16) ? 16 : 0);
  v8f acc = {};
#pragma unroll 4
  for (int k0 = 0; k0 < DH; k0 += 32) {
    __builtin_prefetch(xrow + k0 + 32, 0, 1);
    __builtin_prefetch(wrow + k0 + 32, 0, 1);
    v16h a  = load_afrag_f16(wrow, k0, lane);
    v16h bf = load_bfrag_f16(xrow, k0);
    acc = wmma_f16(a, bf, acc);
  }
#pragma unroll
  for (int i = 0; i < 8; ++i) {
    int o = o0 + i + roff;
    out[((size_t)b * DIM + o) * NTOK + n0 + col] = acc[i] + biasp[o];
  }
}

// ---------- workspace layout (bytes) ----------
static const size_t OFF_XT  = 0;                        // 8*784*384 f16  = 4,816,896
static const size_t OFF_QT  = 4816896;                  // 8*8*784*32 f16 = 3,211,264
static const size_t OFF_KT  = 8028160;                  // same
static const size_t OFF_VN  = 11239424;                 // 8*1024*784 f16 = 12,845,056
static const size_t OFF_VL  = 24084480;                 // 8*1024*784 f32 = 25,690,112
static const size_t OFF_OS  = 49774592;                 // 8*784*1024 f16 = 12,845,056
static const size_t OFF_WH  = 62619648;                 // 1536*384 f16   = 1,179,648
static const size_t OFF_WPH = 63799296;                 // 384*1024 f16   =   786,432
static const size_t OFF_BIA = 64585728;                 // 1536 f32       =     6,144
static const size_t OFF_BIP = 64591872;                 // 384 f32        =     1,536
// total = 64,593,408 bytes

extern "C" void kernel_launch(void* const* d_in, const int* in_sizes, int n_in,
                              void* d_out, int out_size, void* d_ws, size_t ws_size,
                              hipStream_t stream) {
  const float* x    = (const float*)d_in[0];
  const float* Wq   = (const float*)d_in[1];
  const float* bq   = (const float*)d_in[2];
  const float* q_s  = (const float*)d_in[3];
  const float* q_b  = (const float*)d_in[4];
  const float* Wk   = (const float*)d_in[5];
  const float* bk   = (const float*)d_in[6];
  const float* k_s  = (const float*)d_in[7];
  const float* k_b  = (const float*)d_in[8];
  const float* Wv   = (const float*)d_in[9];
  const float* bv   = (const float*)d_in[10];
  const float* v_s  = (const float*)d_in[11];
  const float* v_b  = (const float*)d_in[12];
  const float* Wvl  = (const float*)d_in[13];
  const float* bvl  = (const float*)d_in[14];
  const float* vl_s = (const float*)d_in[15];
  const float* vl_b = (const float*)d_in[16];
  const float* Wth1 = (const float*)d_in[17];
  const float* bth1 = (const float*)d_in[18];
  const float* Wth2 = (const float*)d_in[19];
  const float* bth2 = (const float*)d_in[20];
  const float* Wp   = (const float*)d_in[21];
  const float* bp   = (const float*)d_in[22];
  const float* p_s  = (const float*)d_in[23];
  const float* p_b  = (const float*)d_in[24];
  const float* ab   = (const float*)d_in[25];
  const int*   bidx = (const int*)d_in[26];
  float* out = (float*)d_out;

  char* ws = (char*)d_ws;
  _Float16* xt     = (_Float16*)(ws + OFF_XT);
  _Float16* q_t    = (_Float16*)(ws + OFF_QT);
  _Float16* k_t    = (_Float16*)(ws + OFF_KT);
  _Float16* v_n    = (_Float16*)(ws + OFF_VN);
  float*    vl     = (float*)   (ws + OFF_VL);
  _Float16* osum_t = (_Float16*)(ws + OFF_OS);
  _Float16* wh     = (_Float16*)(ws + OFF_WH);
  _Float16* wph    = (_Float16*)(ws + OFF_WPH);
  float*    bias   = (float*)   (ws + OFF_BIA);
  float*    biasp  = (float*)   (ws + OFF_BIP);

  const size_t attn_smem = (size_t)16 * NTOK * 8 * 2 + (128 + 64 + 64 + 8 + 8) * 4;
  // Unconditional every call: deterministic, idempotent, not a stream op.
  hipFuncSetAttribute((const void*)k_attn,
                      hipFuncAttributeMaxDynamicSharedMemorySize,
                      (int)attn_smem);

  // W: one-shot f16 weight conversion (BN scale folded)
  k_wcvt<<<(1536 * DIM + DIM * DH + 255) / 256, 256, 0, stream>>>(
      Wq, q_s, Wk, k_s, Wv, v_s, Wp, p_s, wh, wph);
  // B: one-shot folded biases
  k_bcvt<<<(1536 + DIM + 255) / 256, 256, 0, stream>>>(
      bq, q_s, q_b, bk, k_s, k_b, bv, v_s, v_b, bp, p_s, p_b, bias, biasp);
  // 0: transpose + f16 convert of x
  k_xt<<<(BATCH * DIM * NTOK + 255) / 256, 256, 0, stream>>>(x, xt);
  // 1: fused QKV projection (8 * 96 * 49 wave-tiles / 8 waves per block)
  k_qkv<<<(BATCH * 96 * 49) / 8, 256, 0, stream>>>(
      xt, wh, bias, q_t, k_t, v_n);
  // 2: depthwise conv branch
  k_dw<<<(BATCH * DH * NTOK + 255) / 256, 256, 0, stream>>>(
      v_n, Wvl, bvl, vl_s, vl_b, vl);
  // 3: fused attention (one WG per (b, 16-row tile); all-head logits in LDS)
  k_attn<<<BATCH * 49, 256, attn_smem, stream>>>(
      q_t, k_t, v_n, Wth1, bth1, Wth2, bth2, ab, bidx, vl, osum_t);
  // 4: output projection (8 * 24 * 49 wave-tiles / 8 waves per block)
  k_proj<<<(BATCH * 24 * 49) / 8, 256, 0, stream>>>(
      osum_t, wph, biasp, out);
}
